// EnhancedCLIP_5660766896364
// MI455X (gfx1250) — compile-verified, matrix operand order checked by reading, not run on hardware
//
#include <hip/hip_runtime.h>
#include <hip/hip_bf16.h>

typedef __attribute__((ext_vector_type(16))) _Float16 v16h;
typedef __attribute__((ext_vector_type(8)))  _Float16 v8h;
typedef __attribute__((ext_vector_type(8)))  float    v8f;

// ---------------------------------------------------------------------------
// Problem constants
// ---------------------------------------------------------------------------
#define CB      4          // batch
#define CH      16
#define CW      16
#define CL      256        // H*W
#define CCH     1024       // C
#define CDI     2048       // D_INNER
#define CDS     32         // D_STATE
#define CDR     64         // DT_RANK
#define CK      4          // K scans
#define CNT     4096       // NTXT

// ---------------------------------------------------------------------------
// Generic f32 -> f16 convert
// ---------------------------------------------------------------------------
__global__ __launch_bounds__(256) void f32_to_f16_kernel(const float* __restrict__ in,
                                                         _Float16* __restrict__ out) {
    int i = blockIdx.x * 256 + threadIdx.x;
    out[i] = (_Float16)in[i];
}

// ---------------------------------------------------------------------------
// Block reduction helper (256 threads)
// ---------------------------------------------------------------------------
__device__ __forceinline__ float block_reduce_256(float v, float* sh) {
    int tid = threadIdx.x;
    sh[tid] = v;
    __syncthreads();
    #pragma unroll
    for (int s = 128; s > 0; s >>= 1) {
        if (tid < s) sh[tid] += sh[tid + s];
        __syncthreads();
    }
    float r = sh[0];
    __syncthreads();
    return r;
}

// ---------------------------------------------------------------------------
// Row-wise L2 normalize (f32 in -> f16 out), cols = 1024
// ---------------------------------------------------------------------------
__global__ __launch_bounds__(256) void l2norm_rows_kernel(const float* __restrict__ in,
                                                          _Float16* __restrict__ out,
                                                          int cols) {
    __shared__ float sh[256];
    long long row = blockIdx.x;
    const float* p = in + row * cols;
    _Float16* q = out + row * cols;
    float ss = 0.f;
    for (int c = threadIdx.x; c < cols; c += 256) {
        float v = p[c];
        ss += v * v;
    }
    float tot = block_reduce_256(ss, sh);
    float inv = rsqrtf(tot);
    for (int c = threadIdx.x; c < cols; c += 256) {
        q[c] = (_Float16)(p[c] * inv);
    }
}

// ---------------------------------------------------------------------------
// WMMA f16 GEMM:  C[m,n] = sum_k A[m,k] * B[n,k]   (A: MxK row-major,
// B: NxK row-major i.e. computes A * B^T), f32 accumulate, batched.
// Bbase = B + (batch % bmod) * strideB.
// Block: 256 threads = 8 waves, block tile 64(M) x 128(N), wave tile 32x32.
// ---------------------------------------------------------------------------
__device__ __forceinline__ v16h load_frag_row(const _Float16* __restrict__ p, int khalf) {
    // CDNA5 ISA 16-bit A/B fragment layout for 16x16x32:
    //   elements 0..7  = K in [8*khalf,     8*khalf + 8)
    //   elements 8..15 = K in [16 + 8*khalf, 16 + 8*khalf + 8)
    union { v16h v; v8h h[2]; } u;
    u.h[0] = *(const v8h*)(p + (khalf << 3));
    u.h[1] = *(const v8h*)(p + 16 + (khalf << 3));
    return u.v;
}

__global__ __launch_bounds__(256) void gemm_f16_wmma_kernel(
    const _Float16* __restrict__ A, const _Float16* __restrict__ Bm,
    float* __restrict__ C,
    int Kd, int lda, int ldb, int ldc,
    long long strideA, long long strideB, long long strideC, int bmod) {
    int batch = blockIdx.z;
    const _Float16* Ab = A + (long long)batch * strideA;
    const _Float16* Bb = Bm + (long long)(batch % bmod) * strideB;
    float* Cb = C + (long long)batch * strideC;

    int lane  = threadIdx.x & 31;
    int wave  = threadIdx.x >> 5;
    int wm    = wave >> 2;          // 0..1
    int wn    = wave & 3;           // 0..3
    int m0    = blockIdx.y * 64 + wm * 32;
    int n0    = blockIdx.x * 128 + wn * 32;
    int r     = lane & 15;
    int khalf = lane >> 4;

    v8f acc[2][2] = {};

    const _Float16* a0 = Ab + (long long)(m0 + r) * lda;
    const _Float16* a1 = a0 + 16LL * lda;
    const _Float16* b0 = Bb + (long long)(n0 + r) * ldb;
    const _Float16* b1 = b0 + 16LL * ldb;

    for (int kk = 0; kk < Kd; kk += 32) {
        __builtin_prefetch(a0 + kk + 64, 0, 0);   // global_prefetch_b8
        __builtin_prefetch(b0 + kk + 64, 0, 0);
        v16h fa0 = load_frag_row(a0 + kk, khalf);
        v16h fa1 = load_frag_row(a1 + kk, khalf);
        v16h fb0 = load_frag_row(b0 + kk, khalf);
        v16h fb1 = load_frag_row(b1 + kk, khalf);
        acc[0][0] = __builtin_amdgcn_wmma_f32_16x16x32_f16(false, fa0, false, fb0,
                                                           (short)0, acc[0][0], false, false);
        acc[0][1] = __builtin_amdgcn_wmma_f32_16x16x32_f16(false, fa0, false, fb1,
                                                           (short)0, acc[0][1], false, false);
        acc[1][0] = __builtin_amdgcn_wmma_f32_16x16x32_f16(false, fa1, false, fb0,
                                                           (short)0, acc[1][0], false, false);
        acc[1][1] = __builtin_amdgcn_wmma_f32_16x16x32_f16(false, fa1, false, fb1,
                                                           (short)0, acc[1][1], false, false);
    }

    // D layout (f32 16x16): VGPR v, lane -> M = v + 8*(lane>>4), N = lane&15
    #pragma unroll
    for (int i = 0; i < 2; i++) {
        #pragma unroll
        for (int j = 0; j < 2; j++) {
            int mbase = m0 + i * 16 + khalf * 8;
            int ncol  = n0 + j * 16 + r;
            #pragma unroll
            for (int v = 0; v < 8; v++) {
                Cb[(long long)(mbase + v) * ldc + ncol] = acc[i][j][v];
            }
        }
    }
}

// ---------------------------------------------------------------------------
// Depthwise 3x3 conv + bias + SiLU.
// xz layout: [b, l=h*16+w, 4096]; x part is e < 2048. Output xc: [b, l, d].
// ---------------------------------------------------------------------------
__global__ __launch_bounds__(256) void conv_dw_silu_kernel(const float* __restrict__ xz,
                                                           const float* __restrict__ cw,
                                                           const float* __restrict__ cb,
                                                           float* __restrict__ xc) {
    int idx = blockIdx.x * 256 + threadIdx.x;   // over B*L*D = 2,097,152
    int d = idx & (CDI - 1);
    int l = (idx >> 11) & (CL - 1);
    int b = idx >> 19;
    int h = l >> 4, w = l & 15;
    float s = 0.f;
    #pragma unroll
    for (int dy = 0; dy < 3; dy++) {
        int hh = h + dy - 1;
        if (hh < 0 || hh >= CH) continue;
        #pragma unroll
        for (int dx = 0; dx < 3; dx++) {
            int ww = w + dx - 1;
            if (ww < 0 || ww >= CW) continue;
            s += xz[((long long)(b * CL + hh * CW + ww)) * (2 * CDI) + d] * cw[d * 9 + dy * 3 + dx];
        }
    }
    s += cb[d];
    s = s / (1.f + __expf(-s));                 // SiLU
    xc[(long long)(b * CL + l) * CDI + d] = s;
}

// ---------------------------------------------------------------------------
// Cross-scan: xs[b,k,l,d] = xc[b, pos(k,l), d] as f16
// ---------------------------------------------------------------------------
__global__ __launch_bounds__(256) void build_xs_kernel(const float* __restrict__ xc,
                                                       _Float16* __restrict__ xs) {
    int idx = blockIdx.x * 256 + threadIdx.x;   // over B*K*L*D = 8,388,608
    int d = idx & (CDI - 1);
    int l = (idx >> 11) & (CL - 1);
    int k = (idx >> 19) & 3;
    int b = idx >> 21;
    int j = (k & 2) ? (CL - 1 - l) : l;                      // reverse for k=2,3
    int pos = (k & 1) ? ((j & 15) * CW + (j >> 4)) : j;      // transpose for k=1,3
    xs[idx] = (_Float16)xc[(long long)(b * CL + pos) * CDI + d];
}

// ---------------------------------------------------------------------------
// Slice dts (first 64 cols of x_dbl) to f16 for the dt_proj GEMM
// ---------------------------------------------------------------------------
__global__ __launch_bounds__(256) void build_dtin_kernel(const float* __restrict__ xdbl,
                                                         _Float16* __restrict__ dtin) {
    int idx = blockIdx.x * 256 + threadIdx.x;   // over (B*K*L)*64 = 262,144
    int r = idx & (CDR - 1);
    int row = idx >> 6;
    dtin[idx] = (_Float16)xdbl[(long long)row * 128 + r];
}

// ---------------------------------------------------------------------------
// Selective scan. One lane per (b,k,d) channel; 32 states + 32 A in VGPRs.
// dtraw: [bk, l, d] pre-softplus; xdbl: [bk, l, 128] (B at +64, C at +96).
// ys out: [bk, l, d] = y + Ds*u.
// ---------------------------------------------------------------------------
__global__ __launch_bounds__(256) void scan_kernel(const _Float16* __restrict__ xs,
                                                   const float* __restrict__ dtraw,
                                                   const float* __restrict__ xdbl,
                                                   const float* __restrict__ dtb,
                                                   const float* __restrict__ A_logs,
                                                   const float* __restrict__ Ds,
                                                   float* __restrict__ ys) {
    int bk = blockIdx.x;                        // 0..15 = b*4+k
    int k = bk & 3;
    int d = blockIdx.y * 256 + threadIdx.x;     // 0..2047

    float Av[CDS];
    #pragma unroll
    for (int n = 0; n < CDS; n++)
        Av[n] = -__expf(A_logs[((long long)(k * CDI + d)) * CDS + n]);

    float bias = dtb[k * CDI + d];
    float Dv = Ds[k * CDI + d];

    float hst[CDS];
    #pragma unroll
    for (int n = 0; n < CDS; n++) hst[n] = 0.f;

    const _Float16* up = xs + (long long)bk * CL * CDI + d;
    const float* dtp = dtraw + (long long)bk * CL * CDI + d;
    const float* bc = xdbl + (long long)bk * CL * 128;
    float* yp = ys + (long long)bk * CL * CDI + d;

    for (int l = 0; l < CL; l++) {
        float u = (float)up[(long long)l * CDI];
        float x = dtp[(long long)l * CDI] + bias;
        float dtv = (x > 20.f) ? x : log1pf(__expf(x));   // softplus
        const float* Bt = bc + l * 128 + CDR;
        const float* Ct = Bt + CDS;
        float du = dtv * u;
        float y = 0.f;
        #pragma unroll
        for (int n = 0; n < CDS; n++) {
            float dA = __expf(dtv * Av[n]);
            hst[n] = fmaf(hst[n], dA, du * Bt[n]);
            y = fmaf(hst[n], Ct[n], y);
        }
        yp[(long long)l * CDI] = y + Dv * u;
    }
}

// ---------------------------------------------------------------------------
// Cross-merge + LayerNorm over channels + SiLU gate -> f16 [b,l,d]
// ---------------------------------------------------------------------------
__global__ __launch_bounds__(256) void merge_norm_gate_kernel(const float* __restrict__ ys,
                                                              const float* __restrict__ xz,
                                                              const float* __restrict__ nw,
                                                              const float* __restrict__ nb,
                                                              _Float16* __restrict__ yout) {
    __shared__ float sh[256];
    int bl = blockIdx.x;                        // 0..1023 = b*256 + l
    int b = bl >> 8;
    int l = bl & 255;
    int h = l >> 4, w = l & 15;
    int lt = w * CH + h;                        // col-major scan index for (h,w)

    const float* y0 = ys + ((long long)(b * 4 + 0) * CL + l) * CDI;
    const float* y1 = ys + ((long long)(b * 4 + 1) * CL + lt) * CDI;
    const float* y2 = ys + ((long long)(b * 4 + 2) * CL + (CL - 1 - l)) * CDI;
    const float* y3 = ys + ((long long)(b * 4 + 3) * CL + (CL - 1 - lt)) * CDI;

    float vals[8];
    float s = 0.f, ss = 0.f;
    #pragma unroll
    for (int i = 0; i < 8; i++) {
        int d = threadIdx.x + i * 256;
        float v = y0[d] + y1[d] + y2[d] + y3[d];
        vals[i] = v;
        s += v;
        ss += v * v;
    }
    float tsum = block_reduce_256(s, sh);
    float tss = block_reduce_256(ss, sh);
    float mu = tsum * (1.f / CDI);
    float var = tss * (1.f / CDI) - mu * mu;
    float inv = rsqrtf(var + 1e-5f);

    const float* zp = xz + (long long)bl * (2 * CDI) + CDI;
    _Float16* op = yout + (long long)bl * CDI;
    #pragma unroll
    for (int i = 0; i < 8; i++) {
        int d = threadIdx.x + i * 256;
        float zn = zp[d];
        float g = zn / (1.f + __expf(-zn));     // SiLU gate
        float o = (vals[i] - mu) * inv * nw[d] + nb[d];
        op[d] = (_Float16)(o * g);
    }
}

// ---------------------------------------------------------------------------
// Host launcher
// ---------------------------------------------------------------------------
extern "C" void kernel_launch(void* const* d_in, const int* in_sizes, int n_in,
                              void* d_out, int out_size, void* d_ws, size_t ws_size,
                              hipStream_t stream) {
    (void)in_sizes; (void)n_in; (void)out_size; (void)ws_size;
    const float* img    = (const float*)d_in[0];   // (4,16,16,1024)
    const float* txt    = (const float*)d_in[1];   // (4096,1024)
    const float* w_in   = (const float*)d_in[2];   // (4096,1024)
    const float* conv_w = (const float*)d_in[3];   // (2048,1,3,3)
    const float* conv_b = (const float*)d_in[4];   // (2048)
    const float* xpw    = (const float*)d_in[5];   // (4,128,2048)
    const float* dtw    = (const float*)d_in[6];   // (4,2048,64)
    const float* dtb    = (const float*)d_in[7];   // (4,2048)
    const float* A_logs = (const float*)d_in[8];   // (8192,32)
    const float* Ds     = (const float*)d_in[9];   // (8192)
    const float* onw    = (const float*)d_in[10];  // (2048)
    const float* onb    = (const float*)d_in[11];  // (2048)
    const float* opw    = (const float*)d_in[12];  // (1024,2048)
    float* out = (float*)d_out;

    char* ws = (char*)d_ws;
    // Arena layout with lifetime-based aliasing (~123 MB total):
    const size_t A1 = 0;                         // 16.8MB: xz f32; later enh + imgn
    const size_t A2 = 16777216;                  // 33.6MB: imgf16 | winf16; later dtraw
    const size_t A3 = 50331648;                  // 33.6MB: xc | dtin; later ys
    const size_t A4 = 83886080;                  // 16.8MB: xs f16
    const size_t A5 = 100663296;                 //  2.1MB: xdbl f32
    const size_t A6 = 102760448;                 // 15.7MB: xpw16|dtw16|opw16|txtn
    const size_t A7 = 118489088;                 //  4.2MB: yf16

    float*    xz     = (float*)(ws + A1);
    float*    enh    = (float*)(ws + A1);                  // after xz dies
    _Float16* imgn   = (_Float16*)(ws + A1 + 4194304);
    _Float16* imgf16 = (_Float16*)(ws + A2);
    _Float16* winf16 = (_Float16*)(ws + A2 + 2097152);
    float*    dtraw  = (float*)(ws + A2);                  // after imgf16/winf16 die
    float*    xc     = (float*)(ws + A3);
    _Float16* dtin   = (_Float16*)(ws + A3 + 8388608);
    float*    ys     = (float*)(ws + A3);                  // after xc/dtin die
    _Float16* xs     = (_Float16*)(ws + A4);
    float*    xdbl   = (float*)(ws + A5);
    _Float16* xpw16  = (_Float16*)(ws + A6);
    _Float16* dtw16  = (_Float16*)(ws + A6 + 2097152);
    _Float16* opw16  = (_Float16*)(ws + A6 + 3145728);
    _Float16* txtn   = (_Float16*)(ws + A6 + 7340032);
    _Float16* yf16   = (_Float16*)(ws + A7);

    // --- converts ---
    f32_to_f16_kernel<<<1048576 / 256, 256, 0, stream>>>(img, imgf16);
    f32_to_f16_kernel<<<4194304 / 256, 256, 0, stream>>>(w_in, winf16);
    f32_to_f16_kernel<<<1048576 / 256, 256, 0, stream>>>(xpw, xpw16);
    f32_to_f16_kernel<<<524288 / 256, 256, 0, stream>>>(dtw, dtw16);
    f32_to_f16_kernel<<<2097152 / 256, 256, 0, stream>>>(opw, opw16);
    l2norm_rows_kernel<<<CNT, 256, 0, stream>>>(txt, txtn, CCH);

    // --- in_proj: (1024x1024)*(4096x1024)^T -> xz (1024x4096) ---
    gemm_f16_wmma_kernel<<<dim3(4096 / 128, 1024 / 64, 1), 256, 0, stream>>>(
        imgf16, winf16, xz, CCH, CCH, CCH, 2 * CDI, 0, 0, 0, 1);

    // --- depthwise conv + SiLU -> xc [b,l,d] ---
    conv_dw_silu_kernel<<<2097152 / 256, 256, 0, stream>>>(xz, conv_w, conv_b, xc);

    // --- cross-scan -> xs f16 [b,k,l,d] ---
    build_xs_kernel<<<8388608 / 256, 256, 0, stream>>>(xc, xs);

    // --- x_proj: 16 batches (256x2048)*(128x2048)^T -> xdbl [bk,l,128] ---
    gemm_f16_wmma_kernel<<<dim3(128 / 128, 256 / 64, 16), 256, 0, stream>>>(
        xs, xpw16, xdbl, CDI, CDI, CDI, 128,
        (long long)CL * CDI, 128LL * CDI, (long long)CL * 128, 4);

    // --- dt slice -> f16 ---
    build_dtin_kernel<<<262144 / 256, 256, 0, stream>>>(xdbl, dtin);

    // --- dt_proj: 16 batches (256x64)*(2048x64)^T -> dtraw [bk,l,2048] ---
    gemm_f16_wmma_kernel<<<dim3(2048 / 128, 256 / 64, 16), 256, 0, stream>>>(
        dtin, dtw16, dtraw, CDR, CDR, CDR, CDI,
        (long long)CL * CDR, (long long)CDI * CDR, (long long)CL * CDI, 4);

    // --- selective scan -> ys [bk,l,d] ---
    scan_kernel<<<dim3(16, CDI / 256), 256, 0, stream>>>(xs, dtraw, xdbl, dtb,
                                                         A_logs, Ds, ys);

    // --- merge + LayerNorm + gate -> yf16 [b,l,d] ---
    merge_norm_gate_kernel<<<CB * CL, 256, 0, stream>>>(ys, xz, onw, onb, yf16);

    // --- out_proj: (1024x2048)*(1024x2048)^T -> enh (1024x1024) ---
    gemm_f16_wmma_kernel<<<dim3(1024 / 128, 1024 / 64, 1), 256, 0, stream>>>(
        yf16, opw16, enh, CDI, CDI, CDI, CCH, 0, 0, 0, 1);

    // --- normalize image rows -> imgn f16 ---
    l2norm_rows_kernel<<<CB * CL, 256, 0, stream>>>(enh, imgn, CCH);

    // --- similarity: (1024x1024)*(4096x1024)^T -> out (1024x4096) ---
    gemm_f16_wmma_kernel<<<dim3(4096 / 128, 1024 / 64, 1), 256, 0, stream>>>(
        imgn, txtn, out, CCH, CCH, CCH, CNT, 0, 0, 0, 1);
}
